// IndependentSatelliteMamba_25589415149575
// MI455X (gfx1250) — compile-verified, matrix operand order checked by reading, not run on hardware
//
#include <hip/hip_runtime.h>
#include <hip/hip_bf16.h>
#include <stdint.h>

// ---------------------------------------------------------------------------
// Problem constants (from the reference)
// ---------------------------------------------------------------------------
#define BN      16
#define WN      512
#define SN      16
#define FN      18
#define DMODEL  256
#define DINNER  512
#define DSTATE  16
#define DTRANK  16
#define DCONV   4
#define NLAYERS 2
#define MROWS   (BN * SN * WN)   // 131072 rows (seq-batch * time)

typedef __attribute__((ext_vector_type(16))) __bf16 v16bf;
typedef __attribute__((ext_vector_type(8)))  float  v8f;
typedef unsigned int u32;

#define USE_ASYNC_LDS 1

// ---------------------------------------------------------------------------
// small helpers
// ---------------------------------------------------------------------------
__device__ __forceinline__ float silu_f(float x) {
  return x / (1.0f + __expf(-x));
}
__device__ __forceinline__ float softplus_f(float x) {
  return x > 0.0f ? x + log1pf(__expf(-x)) : log1pf(__expf(x));
}

#if USE_ASYNC_LDS
// Per-lane async 16-byte copy global -> LDS (ASYNCcnt-tracked DMA path,
// CDNA5 ISA). LDS operand is the wave-relative LDS
// byte address = low 32 bits of the generic shared-memory pointer.
__device__ __forceinline__ void async_cp16(const void* g, void* l) {
  u32 lds = (u32)(size_t)l;
  unsigned long long ga = (unsigned long long)(size_t)g;
  asm volatile("global_load_async_to_lds_b128 %0, %1, off"
               :: "v"(lds), "v"(ga) : "memory");
}
__device__ __forceinline__ void wait_async0() {
#if __has_builtin(__builtin_amdgcn_s_wait_asynccnt)
  __builtin_amdgcn_s_wait_asynccnt(0);
#else
  asm volatile("s_wait_asynccnt 0x0" ::: "memory");
#endif
}
#endif

// ---------------------------------------------------------------------------
// f32 -> bf16 cast kernel (for weights)
// ---------------------------------------------------------------------------
__global__ void cast_f32_bf16_kernel(const float* __restrict__ s,
                                     __bf16* __restrict__ d, int n) {
  int i = blockIdx.x * 256 + threadIdx.x;
  if (i < n) d[i] = (__bf16)s[i];
}

// ---------------------------------------------------------------------------
// Input projection: h[r][d] = x' @ Wp^T + bp   (K = 18, tiny -> VALU kernel)
// ---------------------------------------------------------------------------
__global__ __launch_bounds__(256) void proj_kernel(
    const float* __restrict__ x, const float* __restrict__ Wp,
    const float* __restrict__ bp, float* __restrict__ hF,
    __bf16* __restrict__ hB) {
  int idx = blockIdx.x * 256 + threadIdx.x;        // over MROWS*DMODEL
  int r = idx >> 8, d = idx & 255;
  int bs = r >> 9, w = r & 511;
  int b = bs >> 4, s = bs & 15;
  const float* xp = x + ((size_t)(b * WN + w) * SN + s) * FN;
  const float* wp = Wp + d * FN;
  float acc = bp[d];
#pragma unroll
  for (int f = 0; f < FN; ++f) acc = fmaf(xp[f], wp[f], acc);
  hF[idx] = acc;
  hB[idx] = (__bf16)acc;
}

// ---------------------------------------------------------------------------
// WMMA GEMM:  C[M x N] = A[M x K](bf16) * W[N x K]^T(bf16), f32 accumulate.
// Block tile 128x64, BK=32, 256 threads = 8 waves (4x2), wave tile 32x32
// (four 16x16 WMMA accumulators per wave). Double-buffered LDS with
// GLOBAL_LOAD_ASYNC_TO_LDS_B128 staging: one barrier + one s_wait_asynccnt
// per k-step; the async DMA for tile kt+1 flies under the 4 WMMAs of tile kt.
//
// W rows >= N are staged from a clamped (valid) source row: garbage B-rows n
// only influence output columns n, which the epilogue never stores.
//
// Fragment layout per CDNA5 ISA (CDNA5 ISA, 16-bit A-matrix 16x32):
//   lane L: m = L%16, hi = L/16
//   VGPR j (0..3): K = 8*hi + 2j,2j+1 ; VGPR j (4..7): K = 16 + 8*hi + 2(j-4)..
// B fragment assumed column-major mirror of A (lane indexes N, same K swizzle).
// ---------------------------------------------------------------------------
#define GEMM_MODE_SPLIT 0   // N=1024; n<512 -> outB1(bf16), else -> outB2(bf16)
#define GEMM_MODE_F32   1   // f32 out (row stride N), bounds-checked (N=48)
#define GEMM_MODE_RESID 2   // acc + resid -> resid(f32) and residB(bf16)

#define LDT 40   // LDS row stride (bf16): 80B = 16B-aligned rows for async B128
#define BM  128
#define BNT 64

__device__ __forceinline__ v16bf frag_ld(const __bf16* base) {
  int lane = threadIdx.x & 31;
  int m = lane & 15;
  int hi = (lane >> 4) << 3;   // 0 or 8
  const __bf16* p = base + m * LDT;
  v16bf f;
  u32* fu = (u32*)&f;
#pragma unroll
  for (int j = 0; j < 4; ++j) {
    fu[j]     = *(const u32*)(p + hi + 2 * j);        // K = 8*hi + 2j, +1
    fu[j + 4] = *(const u32*)(p + 16 + hi + 2 * j);   // K = 16 + 8*hi + 2j, +1
  }
  return f;
}

__device__ __forceinline__ void store_acc(
    const v8f& acc, int row0, int col, int N, int mode,
    __bf16* outB1, __bf16* outB2, float* outF, float* resid, __bf16* residB) {
  const float* a = (const float*)&acc;
  if (mode == GEMM_MODE_SPLIT) {
    __bf16* dst = (col < 512) ? outB1 : outB2;
    const int c = (col < 512) ? col : col - 512;
#pragma unroll
    for (int r = 0; r < 8; ++r)
      dst[(size_t)(row0 + r) * 512 + c] = (__bf16)a[r];
  } else if (mode == GEMM_MODE_F32) {
    if (col < N) {
#pragma unroll
      for (int r = 0; r < 8; ++r)
        outF[(size_t)(row0 + r) * N + col] = a[r];
    }
  } else {  // GEMM_MODE_RESID
#pragma unroll
    for (int r = 0; r < 8; ++r) {
      const size_t i = (size_t)(row0 + r) * N + col;
      const float rv = a[r] + resid[i];
      resid[i] = rv;
      residB[i] = (__bf16)rv;
    }
  }
}

__global__ __launch_bounds__(256) void wmma_gemm_kernel(
    const __bf16* __restrict__ A, const __bf16* __restrict__ Wt,
    int M, int N, int K, int mode,
    __bf16* outB1, __bf16* outB2, float* outF,
    float* resid, __bf16* residB) {
  __shared__ __bf16 As[2 * BM * LDT];
  __shared__ __bf16 Bs[2 * BNT * LDT];

  const int tid  = threadIdx.x;
  const int m0   = blockIdx.x * BM;
  const int n0   = blockIdx.y * BNT;
  const int wave = tid >> 5, lane = tid & 31;
  const int wm   = wave & 3;      // 0..3 : 32-row strip
  const int wn   = wave >> 2;     // 0..1 : 32-col strip

  v8f acc00 = {}, acc01 = {}, acc10 = {}, acc11 = {};

  const int li = tid >> 2;          // 0..63 staging row
  const int lk = (tid & 3) * 8;     // 0,8,16,24 staging col (bf16 elems)
  const int wrow = (n0 + li < N) ? (n0 + li) : (N - 1);   // clamped W row

  const int nk = K / 32;

#if USE_ASYNC_LDS
  // ---- prologue: async-stage tile 0 into buffer 0 ----
  {
    const __bf16* ga = A + (size_t)(m0 + li) * K + lk;
    async_cp16(ga, As + li * LDT + lk);
    async_cp16(ga + (size_t)64 * K, As + (li + 64) * LDT + lk);
    async_cp16(Wt + (size_t)wrow * K + lk, Bs + li * LDT + lk);
  }
  wait_async0();
  __syncthreads();

  for (int kt = 0; kt < nk; ++kt) {
    const int cur = kt & 1, nxt = cur ^ 1;
    const bool has_next = (kt + 1 < nk);

    // ---- issue async DMA for next tile (hidden under the WMMAs) ----
    if (has_next) {
      const int k0 = (kt + 1) * 32;
      const __bf16* ga = A + (size_t)(m0 + li) * K + k0 + lk;
      async_cp16(ga, As + (nxt * BM + li) * LDT + lk);
      async_cp16(ga + (size_t)64 * K, As + (nxt * BM + li + 64) * LDT + lk);
      async_cp16(Wt + (size_t)wrow * K + k0 + lk,
                 Bs + (nxt * BNT + li) * LDT + lk);
      if (kt + 2 < nk) __builtin_prefetch(ga + 32, 0, 1);  // global_prefetch_b8
    }

    // ---- compute on current buffer ----
    const __bf16* Ac = As + cur * BM * LDT;
    const __bf16* Bc = Bs + cur * BNT * LDT;
    v16bf fa0 = frag_ld(Ac + (wm * 32) * LDT);
    v16bf fa1 = frag_ld(Ac + (wm * 32 + 16) * LDT);
    v16bf fb0 = frag_ld(Bc + (wn * 32) * LDT);
    v16bf fb1 = frag_ld(Bc + (wn * 32 + 16) * LDT);

    acc00 = __builtin_amdgcn_wmma_f32_16x16x32_bf16(
        false, fa0, false, fb0, (short)0, acc00, false, false);
    acc01 = __builtin_amdgcn_wmma_f32_16x16x32_bf16(
        false, fa0, false, fb1, (short)0, acc01, false, false);
    acc10 = __builtin_amdgcn_wmma_f32_16x16x32_bf16(
        false, fa1, false, fb0, (short)0, acc10, false, false);
    acc11 = __builtin_amdgcn_wmma_f32_16x16x32_bf16(
        false, fa1, false, fb1, (short)0, acc11, false, false);

    if (has_next) wait_async0();
    __syncthreads();
  }
#else
  // ---- fallback: synchronous staging (proven path) ----
  {
    const __bf16* ga = A + (size_t)(m0 + li) * K + lk;
    uint4 va0 = *(const uint4*)ga;
    uint4 va1 = *(const uint4*)(ga + (size_t)64 * K);
    uint4 vb  = *(const uint4*)(Wt + (size_t)wrow * K + lk);
    u32* da0 = (u32*)(As + li * LDT + lk);
    u32* da1 = (u32*)(As + (li + 64) * LDT + lk);
    u32* db  = (u32*)(Bs + li * LDT + lk);
    da0[0] = va0.x; da0[1] = va0.y; da0[2] = va0.z; da0[3] = va0.w;
    da1[0] = va1.x; da1[1] = va1.y; da1[2] = va1.z; da1[3] = va1.w;
    db[0]  = vb.x;  db[1]  = vb.y;  db[2]  = vb.z;  db[3]  = vb.w;
  }
  __syncthreads();
  for (int kt = 0; kt < nk; ++kt) {
    const int cur = kt & 1, nxt = cur ^ 1;
    const bool has_next = (kt + 1 < nk);
    uint4 va0, va1, vb;
    if (has_next) {
      const int k0 = (kt + 1) * 32;
      const __bf16* ga = A + (size_t)(m0 + li) * K + k0 + lk;
      va0 = *(const uint4*)ga;
      va1 = *(const uint4*)(ga + (size_t)64 * K);
      vb  = *(const uint4*)(Wt + (size_t)wrow * K + k0 + lk);
    }
    const __bf16* Ac = As + cur * BM * LDT;
    const __bf16* Bc = Bs + cur * BNT * LDT;
    v16bf fa0 = frag_ld(Ac + (wm * 32) * LDT);
    v16bf fa1 = frag_ld(Ac + (wm * 32 + 16) * LDT);
    v16bf fb0 = frag_ld(Bc + (wn * 32) * LDT);
    v16bf fb1 = frag_ld(Bc + (wn * 32 + 16) * LDT);
    acc00 = __builtin_amdgcn_wmma_f32_16x16x32_bf16(
        false, fa0, false, fb0, (short)0, acc00, false, false);
    acc01 = __builtin_amdgcn_wmma_f32_16x16x32_bf16(
        false, fa0, false, fb1, (short)0, acc01, false, false);
    acc10 = __builtin_amdgcn_wmma_f32_16x16x32_bf16(
        false, fa1, false, fb0, (short)0, acc10, false, false);
    acc11 = __builtin_amdgcn_wmma_f32_16x16x32_bf16(
        false, fa1, false, fb1, (short)0, acc11, false, false);
    if (has_next) {
      u32* da0 = (u32*)(As + (nxt * BM + li) * LDT + lk);
      u32* da1 = (u32*)(As + (nxt * BM + li + 64) * LDT + lk);
      u32* db  = (u32*)(Bs + (nxt * BNT + li) * LDT + lk);
      da0[0] = va0.x; da0[1] = va0.y; da0[2] = va0.z; da0[3] = va0.w;
      da1[0] = va1.x; da1[1] = va1.y; da1[2] = va1.z; da1[3] = va1.w;
      db[0]  = vb.x;  db[1]  = vb.y;  db[2]  = vb.z;  db[3]  = vb.w;
    }
    __syncthreads();
  }
#endif

  // ---- store: C/D f32 layout: VGPR r -> M = 8*(lane/16) + r, N = lane%16 ----
  const int cn = lane & 15;
  const int hi = (lane >> 4) * 8;
  const int rbase = m0 + wm * 32 + hi;
  const int cbase = n0 + wn * 32 + cn;
  store_acc(acc00, rbase,      cbase,      N, mode, outB1, outB2, outF, resid, residB);
  store_acc(acc01, rbase,      cbase + 16, N, mode, outB1, outB2, outF, resid, residB);
  store_acc(acc10, rbase + 16, cbase,      N, mode, outB1, outB2, outF, resid, residB);
  store_acc(acc11, rbase + 16, cbase + 16, N, mode, outB1, outB2, outF, resid, residB);
}

// ---------------------------------------------------------------------------
// Depthwise causal conv (width 4) + bias + SiLU. u layout: [row=bs*WN+w][d]
// ---------------------------------------------------------------------------
__global__ __launch_bounds__(256) void conv_silu_kernel(
    const __bf16* __restrict__ u, const float* __restrict__ Wc,
    const float* __restrict__ bc, __bf16* __restrict__ ua) {
  int idx = blockIdx.x * 256 + threadIdx.x;   // over MROWS*DINNER
  int r = idx >> 9, d = idx & 511;
  int bs = r >> 9, w = r & 511;
  float acc = bc[d];
#pragma unroll
  for (int j = 0; j < DCONV; ++j) {
    int t = w - (DCONV - 1) + j;
    if (t >= 0) {
      float uv = (float)u[((size_t)(bs << 9) + t) * DINNER + d];
      acc = fmaf(uv, Wc[d * DCONV + j], acc);
    }
  }
  ua[idx] = (__bf16)silu_f(acc);
}

// ---------------------------------------------------------------------------
// dt = softplus(dt_r @ Wdt^T + bdt), K = DT_RANK = 16 (tiny -> VALU kernel)
// ---------------------------------------------------------------------------
__global__ __launch_bounds__(256) void dt_kernel(
    const float* __restrict__ xdbl, const float* __restrict__ Wdt,
    const float* __restrict__ bdt, __bf16* __restrict__ dt) {
  int idx = blockIdx.x * 256 + threadIdx.x;   // over MROWS*DINNER
  int r = idx >> 9, d = idx & 511;
  const float* xr = xdbl + (size_t)r * (DTRANK + 2 * DSTATE);
  const float* wr = Wdt + d * DTRANK;
  float acc = bdt[d];
#pragma unroll
  for (int k = 0; k < DTRANK; ++k) acc = fmaf(xr[k], wr[k], acc);
  dt[idx] = (__bf16)softplus_f(acc);
}

// ---------------------------------------------------------------------------
// Selective scan. One thread per (sequence bs, channel d); 512 steps, 16
// states in registers. B/C staged through LDS in chunks of 16 timesteps.
// Fused epilogue: y = (scan + u*D) * silu(z); y aliases z (per-thread RMW).
// ---------------------------------------------------------------------------
#define SCHUNK 16

__global__ __launch_bounds__(128) void scan_kernel(
    const __bf16* __restrict__ ua, const __bf16* __restrict__ dtv,
    const __bf16* z, const float* __restrict__ xdbl,
    const float* __restrict__ A_log, const float* __restrict__ Dskip,
    __bf16* y) {
  __shared__ float bcs[SCHUNK][2 * DSTATE];
  const int bs = blockIdx.y;
  const int d  = blockIdx.x * 128 + threadIdx.x;

  float A[DSTATE];
#pragma unroll
  for (int n = 0; n < DSTATE; ++n) A[n] = -__expf(A_log[d * DSTATE + n]);
  const float Dv = Dskip[d];

  float s[DSTATE];
#pragma unroll
  for (int n = 0; n < DSTATE; ++n) s[n] = 0.0f;

  const size_t rowbase = (size_t)bs * WN;
  for (int t0 = 0; t0 < WN; t0 += SCHUNK) {
    __syncthreads();
#pragma unroll
    for (int i = 0; i < (SCHUNK * 2 * DSTATE) / 128; ++i) {
      int idx = threadIdx.x + i * 128;
      int tt = idx >> 5, nn = idx & 31;
      bcs[tt][nn] =
          xdbl[(rowbase + t0 + tt) * (DTRANK + 2 * DSTATE) + DTRANK + nn];
    }
    __syncthreads();

    for (int tt = 0; tt < SCHUNK; ++tt) {
      const size_t r = rowbase + t0 + tt;
      const float dt = (float)dtv[r * DINNER + d];
      const float uu = (float)ua[r * DINNER + d];
      const float du = dt * uu;
      float yt = 0.0f;
#pragma unroll
      for (int n = 0; n < DSTATE; ++n) {
        s[n] = __expf(dt * A[n]) * s[n] + du * bcs[tt][n];
        yt = fmaf(s[n], bcs[tt][DSTATE + n], yt);
      }
      const float zz = (float)z[r * DINNER + d];
      y[r * DINNER + d] = (__bf16)((yt + uu * Dv) * silu_f(zz));
    }
  }
}

// ---------------------------------------------------------------------------
// Head: last token -> LayerNorm -> 256->128 ReLU -> 128->1, mask.
// One block (256 threads) per sequence.
// ---------------------------------------------------------------------------
__global__ __launch_bounds__(256) void head_kernel(
    const float* __restrict__ hF, const float* __restrict__ g,
    const float* __restrict__ beta, const float* __restrict__ W1,
    const float* __restrict__ b1, const float* __restrict__ W2,
    const float* __restrict__ b2, const unsigned char* __restrict__ mask,
    float* __restrict__ out) {
  __shared__ float red[256];
  __shared__ float ln[DMODEL];
  __shared__ float hid[DMODEL / 2];
  const int bs = blockIdx.x, tid = threadIdx.x;

  const float v = hF[((size_t)bs * WN + (WN - 1)) * DMODEL + tid];

  red[tid] = v;
  __syncthreads();
  for (int off = 128; off > 0; off >>= 1) {
    if (tid < off) red[tid] += red[tid + off];
    __syncthreads();
  }
  const float mu = red[0] * (1.0f / DMODEL);
  __syncthreads();

  const float dv = v - mu;
  red[tid] = dv * dv;
  __syncthreads();
  for (int off = 128; off > 0; off >>= 1) {
    if (tid < off) red[tid] += red[tid + off];
    __syncthreads();
  }
  const float var = red[0] * (1.0f / DMODEL);

  ln[tid] = dv * rsqrtf(var + 1e-5f) * g[tid] + beta[tid];
  __syncthreads();

  if (tid < DMODEL / 2) {
    const float* w = W1 + tid * DMODEL;
    float a = b1[tid];
#pragma unroll 8
    for (int k = 0; k < DMODEL; ++k) a = fmaf(ln[k], w[k], a);
    hid[tid] = fmaxf(a, 0.0f);
  }
  __syncthreads();

  red[tid] = (tid < DMODEL / 2) ? hid[tid] * W2[tid] : 0.0f;
  __syncthreads();
  for (int off = 128; off > 0; off >>= 1) {
    if (tid < off) red[tid] += red[tid + off];
    __syncthreads();
  }
  if (tid == 0) out[bs] = mask[bs] ? (red[0] + b2[0]) : 0.0f;
}

// ---------------------------------------------------------------------------
// Host orchestration
// ---------------------------------------------------------------------------
extern "C" void kernel_launch(void* const* d_in, const int* in_sizes, int n_in,
                              void* d_out, int out_size, void* d_ws, size_t ws_size,
                              hipStream_t stream) {
  const float* x      = (const float*)d_in[0];
  const unsigned char* mask = (const unsigned char*)d_in[1];
  const float* Wp     = (const float*)d_in[2];
  const float* bp     = (const float*)d_in[3];
  const float* Win    = (const float*)d_in[4];
  const float* Wconv  = (const float*)d_in[5];
  const float* bconv  = (const float*)d_in[6];
  const float* Wx     = (const float*)d_in[7];
  const float* Wdt    = (const float*)d_in[8];
  const float* bdt    = (const float*)d_in[9];
  const float* A_log  = (const float*)d_in[10];
  const float* Dskip  = (const float*)d_in[11];
  const float* Wout   = (const float*)d_in[12];
  const float* ln_g   = (const float*)d_in[13];
  const float* ln_b   = (const float*)d_in[14];
  const float* W1     = (const float*)d_in[15];
  const float* b1     = (const float*)d_in[16];
  const float* W2     = (const float*)d_in[17];
  const float* b2     = (const float*)d_in[18];
  float* out = (float*)d_out;

  const size_t M = MROWS;
  char* ws = (char*)d_ws;
  size_t off = 0;
  auto take = [&](size_t bytes) -> char* {
    char* p = ws + off;
    off += (bytes + 255) & ~(size_t)255;
    return p;
  };
  float*  hF   = (float*)take(M * DMODEL * 4);            // fp32 residual stream
  __bf16* hB   = (__bf16*)take(M * DMODEL * 2);           // bf16 copy (GEMM in)
  __bf16* uB   = (__bf16*)take(M * DINNER * 2);           // raw u; reused as dt
  __bf16* zB   = (__bf16*)take(M * DINNER * 2);           // z; reused as y
  __bf16* uaB  = (__bf16*)take(M * DINNER * 2);           // conv+silu(u)
  float*  xdbl = (float*)take(M * (DTRANK + 2 * DSTATE) * 4);
  __bf16* WinB  = (__bf16*)take((size_t)2 * DINNER * DMODEL * 2);
  __bf16* WxB   = (__bf16*)take((size_t)(DTRANK + 2 * DSTATE) * DINNER * 2);
  __bf16* WoutB = (__bf16*)take((size_t)DMODEL * DINNER * 2);

  // ---- input projection ----
  proj_kernel<<<(M * DMODEL) / 256, 256, 0, stream>>>(x, Wp, bp, hF, hB);

  for (int layer = 0; layer < NLAYERS; ++layer) {
    const float* Win_l   = Win   + (size_t)layer * 2 * DINNER * DMODEL;
    const float* Wconv_l = Wconv + (size_t)layer * DINNER * DCONV;
    const float* bconv_l = bconv + (size_t)layer * DINNER;
    const float* Wx_l    = Wx    + (size_t)layer * (DTRANK + 2 * DSTATE) * DINNER;
    const float* Wdt_l   = Wdt   + (size_t)layer * DINNER * DTRANK;
    const float* bdt_l   = bdt   + (size_t)layer * DINNER;
    const float* Alog_l  = A_log + (size_t)layer * DINNER * DSTATE;
    const float* Dsk_l   = Dskip + (size_t)layer * DINNER;
    const float* Wout_l  = Wout  + (size_t)layer * DMODEL * DINNER;

    // xz = h @ Win^T  (M x 1024, K=256), split into u / z (bf16)
    {
      int n = 2 * DINNER * DMODEL;
      cast_f32_bf16_kernel<<<(n + 255) / 256, 256, 0, stream>>>(Win_l, WinB, n);
    }
    wmma_gemm_kernel<<<dim3(M / BM, (2 * DINNER) / BNT), 256, 0, stream>>>(
        hB, WinB, (int)M, 2 * DINNER, DMODEL, GEMM_MODE_SPLIT,
        uB, zB, nullptr, nullptr, nullptr);

    // u = silu(depthwise_conv(u) + bconv)
    conv_silu_kernel<<<(M * DINNER) / 256, 256, 0, stream>>>(uB, Wconv_l,
                                                             bconv_l, uaB);

    // x_dbl = u @ Wx^T  (M x 48, K=512) -> fp32
    {
      int n = (DTRANK + 2 * DSTATE) * DINNER;
      cast_f32_bf16_kernel<<<(n + 255) / 256, 256, 0, stream>>>(Wx_l, WxB, n);
    }
    wmma_gemm_kernel<<<dim3(M / BM, 1), 256, 0, stream>>>(
        uaB, WxB, (int)M, DTRANK + 2 * DSTATE, DINNER, GEMM_MODE_F32,
        nullptr, nullptr, xdbl, nullptr, nullptr);

    // dt = softplus(dt_r @ Wdt^T + bdt)  (reuse raw-u buffer)
    __bf16* dtB = uB;
    dt_kernel<<<(M * DINNER) / 256, 256, 0, stream>>>(xdbl, Wdt_l, bdt_l, dtB);

    // selective scan + fused y = (ys + u*D) * silu(z)   (y reuses z buffer)
    __bf16* yB = zB;
    scan_kernel<<<dim3(DINNER / 128, BN * SN), 128, 0, stream>>>(
        uaB, dtB, zB, xdbl, Alog_l, Dsk_l, yB);

    // h = y @ Wout^T + h  (M x 256, K=512), residual fused
    {
      int n = DMODEL * DINNER;
      cast_f32_bf16_kernel<<<(n + 255) / 256, 256, 0, stream>>>(Wout_l, WoutB, n);
    }
    wmma_gemm_kernel<<<dim3(M / BM, DMODEL / BNT), 256, 0, stream>>>(
        yB, WoutB, (int)M, DMODEL, DINNER, GEMM_MODE_RESID,
        nullptr, nullptr, nullptr, hF, hB);
  }

  // ---- LN + MLP head on last token of each sequence ----
  head_kernel<<<BN * SN, 256, 0, stream>>>(hF, ln_g, ln_b, W1, b1, W2, b2,
                                           mask, out);
}
